// DFMSDABlock_46608985096583
// MI455X (gfx1250) — matrix operand order repaired and checked by reading, not compiled, not run on hardware
//
#include <hip/hip_runtime.h>

// ---------------------------------------------------------------------------
// MI455X (gfx1250) implementation of the DFMSDA block.
//
// Roofline: 8 GEMMs of [16384x384]x[384x384] = ~38.7 GFLOP dominate; total
// HBM traffic ~150MB (~6.5us @ 23.3 TB/s), so the GEMMs ride the WMMA units
// (v_wmma_f32_16x16x32_bf16, f32 accumulate). Tiles are staged into LDS with
// the CDNA5 async copy path (global_load_async_to_lds_b128 / ASYNCcnt),
// weights are pre-converted to bf16 once. Attention (9-tap dilated window,
// 64-dim heads) is <3% of FLOPs -> VALU with bf16 loads.
// ---------------------------------------------------------------------------

typedef __attribute__((ext_vector_type(16))) __bf16       v16bf;
typedef __attribute__((ext_vector_type(8)))  float        v8f;
typedef __attribute__((ext_vector_type(4)))  unsigned int uint4v;

union BF16Frag { v16bf v; uint4v q[2]; };

__device__ __forceinline__ unsigned short f32_to_bf16(float f) {
  unsigned int u = __float_as_uint(f);
  unsigned int r = u + 0x7FFFu + ((u >> 16) & 1u);   // round-to-nearest-even
  return (unsigned short)(r >> 16);
}
__device__ __forceinline__ float bf16lo(unsigned int u) {
  return __uint_as_float(u << 16);
}
__device__ __forceinline__ float bf16hi(unsigned int u) {
  return __uint_as_float(u & 0xFFFF0000u);
}

// CDNA5 async copy: 16B per lane, global -> LDS, tracked by ASYNCcnt.
// lds_off = wave-relative LDS byte address (low 32 bits of generic pointer).
__device__ __forceinline__ void async_load16(unsigned lds_off, const void* gptr) {
  asm volatile("global_load_async_to_lds_b128 %0, %1, off"
               :: "v"(lds_off), "v"(gptr) : "memory");
}
__device__ __forceinline__ void wait_async0() {
  asm volatile("s_wait_asynccnt 0x0" ::: "memory");
}

// dims
#define NB   4
#define NC   384
#define HH   64
#define WW   64
#define NPIX 4096          // HH*WW
#define MROW 16384         // NB*NPIX
#define NELT (MROW * NC)   // 6291456
#define WELT (NC * NC)     // 147456 per weight matrix

// ---------------------------------------------------------------------------
// 0) One-time weight conversion fp32 -> bf16.
// ---------------------------------------------------------------------------
__global__ __launch_bounds__(256)
void wcvt_kernel(const float* __restrict__ src, unsigned short* __restrict__ dst)
{
  int i = blockIdx.x * 256 + threadIdx.x;
  if (i < WELT) dst[i] = f32_to_bf16(src[i]);
}

// ---------------------------------------------------------------------------
// 1) LayerNorm both streams + s1 = vi_f - ir_f. One wave32 per pixel.
//    Inputs are [B,C,H,W]; normalization is over C.
// ---------------------------------------------------------------------------
__global__ __launch_bounds__(256)
void ln_diff_kernel(const float* __restrict__ vi, const float* __restrict__ ir,
                    const float* __restrict__ gvi, const float* __restrict__ bvi,
                    const float* __restrict__ gir, const float* __restrict__ bir,
                    float* __restrict__ s1f,
                    unsigned short* __restrict__ vib, unsigned short* __restrict__ irb,
                    unsigned short* __restrict__ s1b, unsigned short* __restrict__ s2b)
{
  int wave = (blockIdx.x * 256 + threadIdx.x) >> 5;   // pixel id 0..16383
  int lane = threadIdx.x & 31;
  int b = wave >> 12, pix = wave & 4095;
  const float* vp = vi + (size_t)b * NC * NPIX + pix;
  const float* ip = ir + (size_t)b * NC * NPIX + pix;

  float va[12], ia[12];
  float sv = 0.f, qv = 0.f, si = 0.f, qi = 0.f;
#pragma unroll
  for (int j = 0; j < 12; j++) {
    int c = lane + 32 * j;
    float a = vp[(size_t)c * NPIX];
    float d = ip[(size_t)c * NPIX];
    va[j] = a; ia[j] = d;
    sv += a; qv += a * a; si += d; qi += d * d;
  }
#pragma unroll
  for (int off = 16; off >= 1; off >>= 1) {
    sv += __shfl_xor(sv, off, 32);
    qv += __shfl_xor(qv, off, 32);
    si += __shfl_xor(si, off, 32);
    qi += __shfl_xor(qi, off, 32);
  }
  const float inv_n = 1.0f / (float)NC;
  float mv = sv * inv_n, mi_ = si * inv_n;
  float rv = rsqrtf(qv * inv_n - mv * mv + 1e-6f);
  float ri = rsqrtf(qi * inv_n - mi_ * mi_ + 1e-6f);

  size_t base = (size_t)wave * NC;
#pragma unroll
  for (int j = 0; j < 12; j++) {
    int c = lane + 32 * j;
    float nv = (va[j] - mv)  * rv * gvi[c] + bvi[c];
    float ni = (ia[j] - mi_) * ri * gir[c] + bir[c];
    float s  = nv - ni;
    vib[base + c] = f32_to_bf16(nv);
    irb[base + c] = f32_to_bf16(ni);
    s1f[base + c] = s;
    s1b[base + c] = f32_to_bf16(s);
    s2b[base + c] = f32_to_bf16(-s);
  }
}

// ---------------------------------------------------------------------------
// 2) GEMM  D[M,N] = A[M,K] * W[N,K]^T + bias   (K = N = 384, bf16 x bf16)
//    Workgroup: 256 threads = 8 waves (4M x 2N), tile 128x128, K-step 32.
//    Tiles staged with global_load_async_to_lds_b128 (ASYNCcnt).
//    mode 0: store bf16 to Obf[M,384]
//    mode 1: +resSign*resid[M,384], store fp32 transposed to Ofp[B,C,H,W]
// ---------------------------------------------------------------------------
#define TM 128
#define TN 128
#define TK 32

__global__ __launch_bounds__(256)
void gemm_bf16_wmma(const unsigned short* __restrict__ A,
                    const unsigned short* __restrict__ Wb,   // bf16 [N,K]
                    const float* __restrict__ bias,
                    unsigned short* __restrict__ Obf,
                    float* __restrict__ Ofp,
                    const float* __restrict__ resid,
                    float resSign, int mode)
{
  __shared__ unsigned short As[TM][TK];   // 8 KB
  __shared__ unsigned short Bs[TN][TK];   // 8 KB, Bs[n][k]

  const int tid  = threadIdx.x;
  const int lane = tid & 31;
  const int wid  = tid >> 5;
  const int wm   = wid >> 1;              // 0..3
  const int wn   = wid & 1;               // 0..1
  const int m0   = blockIdx.y * TM;
  const int n0   = blockIdx.x * TN;
  const int halfsel = lane >> 4;          // which 16-lane half
  const int l16     = lane & 15;

  const unsigned asbase = (unsigned)(uintptr_t)&As[0][0];
  const unsigned bsbase = (unsigned)(uintptr_t)&Bs[0][0];

  v8f acc[2][4];
#pragma unroll
  for (int i = 0; i < 2; i++)
#pragma unroll
    for (int j = 0; j < 4; j++) acc[i][j] = v8f{};

  for (int k0 = 0; k0 < NC; k0 += TK) {
    if (k0 + TK < NC)  // prefetch next A K-slice (global_prefetch_b8)
      __builtin_prefetch(&A[(size_t)(m0 + (tid >> 4)) * NC + k0 + TK], 0, 1);

    // ---- async-stage A and B tiles: 512 x 16B chunks each, 2 per thread ----
#pragma unroll
    for (int h = 0; h < 2; h++) {
      int c   = tid + h * 256;            // chunk 0..511
      int row = c >> 2;                   // 4 chunks (64B) per 32-elem row
      int kc  = (c & 3) * 8;
      async_load16(asbase + c * 16, A  + (size_t)(m0 + row) * NC + k0 + kc);
      async_load16(bsbase + c * 16, Wb + (size_t)(n0 + row) * NC + k0 + kc);
    }
    wait_async0();
    __syncthreads();

    // ---- assemble fragments per CDNA5 16-bit WMMA VGPR layouts ----
    BF16Frag a[2], b[4];
#pragma unroll
    for (int mi = 0; mi < 2; mi++) {
      int row = wm * 32 + mi * 16 + l16;
      const uint4v* p = (const uint4v*)&As[row][0];
      a[mi].q[0] = p[halfsel];            // K = half*8 .. +7
      a[mi].q[1] = p[2 + halfsel];        // K = 16+half*8 .. +7
    }
#pragma unroll
    for (int ni = 0; ni < 4; ni++) {
      int col = wn * 64 + ni * 16 + l16;
      const uint4v* p = (const uint4v*)&Bs[col][0];
      b[ni].q[0] = p[halfsel];            // K = half*8 .. +7
      b[ni].q[1] = p[2 + halfsel];        // K = 16+half*8 .. +7
    }
#pragma unroll
    for (int mi = 0; mi < 2; mi++)
#pragma unroll
      for (int ni = 0; ni < 4; ni++)
        acc[mi][ni] = __builtin_amdgcn_wmma_f32_16x16x32_bf16(
            false, a[mi].v, false, b[ni].v, (short)0, acc[mi][ni],
            false, false);
    __syncthreads();
  }

  // ---- epilogue: D element (lane L, vgpr j) -> M = j + (L/16)*8, N = L%16
  const int rowbase = halfsel * 8;
#pragma unroll
  for (int mi = 0; mi < 2; mi++)
#pragma unroll
    for (int ni = 0; ni < 4; ni++) {
      int gn = n0 + wn * 64 + ni * 16 + l16;
      float bv = bias[gn];
#pragma unroll
      for (int j = 0; j < 8; j++) {
        int gm = m0 + wm * 32 + mi * 16 + rowbase + j;
        float val = acc[mi][ni][j] + bv;
        if (mode == 0) {
          Obf[(size_t)gm * NC + gn] = f32_to_bf16(val);
        } else {
          val += resSign * resid[(size_t)gm * NC + gn];
          int bb = gm >> 12, pix = gm & 4095;
          Ofp[((size_t)bb * NC + gn) * NPIX + pix] = val;   // NHWC -> NCHW
        }
      }
    }
}

// ---------------------------------------------------------------------------
// 3) Per-pixel dilated-window attention. One thread per (pixel, dil, head).
//    Zero-padded taps contribute logit 0 to the softmax denominator
//    (nn.Unfold zero padding), but v=0 -> no numerator contribution.
// ---------------------------------------------------------------------------
__device__ __forceinline__ float dot8(uint4v a, uint4v b) {
  float s = 0.f;
#pragma unroll
  for (int i = 0; i < 4; i++) {
    unsigned int ua = a[i], ub = b[i];
    s += bf16lo(ua) * bf16lo(ub) + bf16hi(ua) * bf16hi(ub);
  }
  return s;
}

__global__ __launch_bounds__(256)
void attn_kernel(const unsigned short* __restrict__ Q,
                 const unsigned short* __restrict__ Kt,
                 const unsigned short* __restrict__ V,
                 unsigned short* __restrict__ O)
{
  int site = blockIdx.x * 256 + threadIdx.x;   // 98304 sites
  int hh = site & 1;
  int dd = (site >> 1) % 3;
  int m  = site / 6;
  int b = m >> 12, pix = m & 4095, y = pix >> 6, x = pix & 63;
  const int dil = dd + 1;
  const int cb  = dd * 128 + hh * 64;

  const uint4v* q4 = (const uint4v*)(Q + (size_t)m * NC + cb);
  uint4v qv[8];
#pragma unroll
  for (int i = 0; i < 8; i++) qv[i] = q4[i];

  float logit[9];
  bool  valid[9];
#pragma unroll
  for (int t = 0; t < 9; t++) {
    int ny = y + (t / 3 - 1) * dil;
    int nx = x + (t % 3 - 1) * dil;
    bool ok = ((unsigned)ny < (unsigned)HH) && ((unsigned)nx < (unsigned)WW);
    valid[t] = ok;
    float d = 0.f;
    if (ok) {
      const uint4v* k4 =
          (const uint4v*)(Kt + ((size_t)(b * NPIX + ny * WW + nx)) * NC + cb);
#pragma unroll
      for (int i = 0; i < 8; i++) d += dot8(qv[i], k4[i]);
    }
    logit[t] = ok ? d * 0.125f : 0.0f;   // SCALE = 64^-0.5
  }
  float mx = logit[0];
#pragma unroll
  for (int t = 1; t < 9; t++) mx = fmaxf(mx, logit[t]);
  float e[9], sum = 0.f;
#pragma unroll
  for (int t = 0; t < 9; t++) { e[t] = __expf(logit[t] - mx); sum += e[t]; }
  float inv = 1.0f / sum;

  float acc[64];
#pragma unroll
  for (int i = 0; i < 64; i++) acc[i] = 0.f;
#pragma unroll
  for (int t = 0; t < 9; t++) {
    if (!valid[t]) continue;
    int ny = y + (t / 3 - 1) * dil;
    int nx = x + (t % 3 - 1) * dil;
    const uint4v* v4 =
        (const uint4v*)(V + ((size_t)(b * NPIX + ny * WW + nx)) * NC + cb);
    float wt = e[t] * inv;
#pragma unroll
    for (int i = 0; i < 8; i++) {
      uint4v vv = v4[i];
#pragma unroll
      for (int jj = 0; jj < 4; jj++) {
        acc[i * 8 + jj * 2 + 0] += wt * bf16lo(vv[jj]);
        acc[i * 8 + jj * 2 + 1] += wt * bf16hi(vv[jj]);
      }
    }
  }
  unsigned int* op = (unsigned int*)(O + (size_t)m * NC + cb);
#pragma unroll
  for (int i = 0; i < 32; i++) {
    unsigned int lo = f32_to_bf16(acc[2 * i]);
    unsigned int hi = f32_to_bf16(acc[2 * i + 1]);
    op[i] = lo | (hi << 16);
  }
}

// ---------------------------------------------------------------------------
// 4) The reference's "faithful to source" permutation:
//    swapaxes(1,2).reshape per dilation, then stack.transpose(1,2,3,0,4)
//    .reshape(B,H,W,C). Composed into a single gather.
// ---------------------------------------------------------------------------
__global__ __launch_bounds__(256)
void scramble_kernel(const unsigned short* __restrict__ Ain,
                     unsigned short* __restrict__ X)
{
  int idx = blockIdx.x * 256 + threadIdx.x;          // NELT total
  int b = idx / (NPIX * NC);
  int g = idx - b * (NPIX * NC);                     // (y*64+x)*384 + cc
  int x0   = g & 63;
  int t    = g >> 6;
  int dd   = t % 3;
  int u    = t / 3;
  int y0   = u & 63;
  int d128 = u >> 6;                                 // 0..127
  int f  = (d128 << 12) + (y0 << 6) + x0;
  int p  = f >> 7;                                   // source pixel
  int hh = (f >> 6) & 1;
  int e  = f & 63;
  X[idx] = Ain[((size_t)(b * NPIX + p)) * NC + dd * 128 + hh * 64 + e];
}

// ---------------------------------------------------------------------------
extern "C" void kernel_launch(void* const* d_in, const int* in_sizes, int n_in,
                              void* d_out, int out_size, void* d_ws, size_t ws_size,
                              hipStream_t stream)
{
  (void)in_sizes; (void)n_in; (void)out_size; (void)ws_size;
  const float* vi  = (const float*)d_in[0];
  const float* ir  = (const float*)d_in[1];
  const float* gvi = (const float*)d_in[2];
  const float* bvi = (const float*)d_in[3];
  const float* gir = (const float*)d_in[4];
  const float* bir = (const float*)d_in[5];

  const size_t E = (size_t)NELT;
  char* ws = (char*)d_ws;
  float*          s1f = (float*)ws;           ws += E * 4;
  unsigned short* vib = (unsigned short*)ws;  ws += E * 2;
  unsigned short* irb = (unsigned short*)ws;  ws += E * 2;
  unsigned short* s1b = (unsigned short*)ws;  ws += E * 2;
  unsigned short* s2b = (unsigned short*)ws;  ws += E * 2;
  unsigned short* qb  = (unsigned short*)ws;  ws += E * 2;
  unsigned short* kb  = (unsigned short*)ws;  ws += E * 2;
  unsigned short* vb  = (unsigned short*)ws;  ws += E * 2;
  unsigned short* ab  = (unsigned short*)ws;  ws += E * 2;
  unsigned short* xb  = (unsigned short*)ws;  ws += E * 2;
  unsigned short* wbf = (unsigned short*)ws;  ws += (size_t)8 * WELT * 2;

  // bf16 weight copies: order q1,k1,v1,p1,q2,k2,v2,p2 (inputs 6,8,..,20)
  const int wblocks = (WELT + 255) / 256;
  for (int i = 0; i < 8; i++)
    wcvt_kernel<<<wblocks, 256, 0, stream>>>((const float*)d_in[6 + 2 * i],
                                             wbf + (size_t)i * WELT);

  ln_diff_kernel<<<MROW / 8, 256, 0, stream>>>(vi, ir, gvi, bvi, gir, bir,
                                               s1f, vib, irb, s1b, s2b);

  const dim3 ggemm(NC / TN, MROW / TM);   // (3, 128)
  for (int s = 0; s < 2; s++) {
    const unsigned short* wq = wbf + (size_t)(4 * s + 0) * WELT;
    const unsigned short* wk = wbf + (size_t)(4 * s + 1) * WELT;
    const unsigned short* wv = wbf + (size_t)(4 * s + 2) * WELT;
    const unsigned short* wp = wbf + (size_t)(4 * s + 3) * WELT;
    const int o = 6 + s * 8;
    const float* bq = (const float*)d_in[o + 1];
    const float* bk = (const float*)d_in[o + 3];
    const float* bv = (const float*)d_in[o + 5];
    const float* bp = (const float*)d_in[o + 7];

    const unsigned short* Aq  = s ? s2b : s1b;   // s2 = -s1
    const unsigned short* Akv = s ? irb : vib;

    gemm_bf16_wmma<<<ggemm, 256, 0, stream>>>(Aq,  wq, bq, qb, nullptr, nullptr, 0.f, 0);
    gemm_bf16_wmma<<<ggemm, 256, 0, stream>>>(Akv, wk, bk, kb, nullptr, nullptr, 0.f, 0);
    gemm_bf16_wmma<<<ggemm, 256, 0, stream>>>(Akv, wv, bv, vb, nullptr, nullptr, 0.f, 0);

    attn_kernel<<<(MROW * 6) / 256, 256, 0, stream>>>(qb, kb, vb, ab);
    scramble_kernel<<<(int)(E / 256), 256, 0, stream>>>(ab, xb);

    float* outp = (float*)d_out + s * E;
    gemm_bf16_wmma<<<ggemm, 256, 0, stream>>>(xb, wp, bp, nullptr, outp, s1f,
                                              s ? -1.0f : 1.0f, 1);
  }
}
